// sem_model_9440338117403
// MI455X (gfx1250) — matrix operand order repaired and checked by reading, not compiled, not run on hardware
//
#include <hip/hip_runtime.h>
#include <hip/hip_fp16.h>

typedef _Float16 half_t;
typedef __attribute__((ext_vector_type(16))) _Float16 v16h;
typedef __attribute__((ext_vector_type(8)))  _Float16 v8h;
typedef __attribute__((ext_vector_type(8)))  float    v8f;

#define B_   16
#define N1_  16384
#define Q1_  256
#define Q2_  64
#define S_   16

// ---------------------------------------------------------------------------
// Farthest point sampling: one block per batch. Matches jax scan semantics:
// emits `far` BEFORE the min/argmax update; argmax tie-break = lowest index.
// ---------------------------------------------------------------------------
__global__ void fps_kernel(const float* __restrict__ xyz, int N, int npoint,
                           int* __restrict__ out_idx, float* __restrict__ dist) {
  const int b = blockIdx.x;
  const int tid = threadIdx.x;
  const int T = blockDim.x;
  const float* px = xyz + (size_t)b * N * 3;
  float* pd = dist + (size_t)b * N;
  __shared__ float s_val[256];
  __shared__ int   s_idx[256];
  __shared__ int   s_far;
  if (tid == 0) s_far = 0;
  for (int n = tid; n < N; n += T) pd[n] = 1e10f;
  __syncthreads();
  for (int it = 0; it < npoint; ++it) {
    const int far = s_far;
    if (tid == 0) out_idx[b * npoint + it] = far;
    const float cx = px[far * 3 + 0], cy = px[far * 3 + 1], cz = px[far * 3 + 2];
    float bestv = -1.0f; int besti = 0;
    for (int n = tid; n < N; n += T) {
      const float dx = px[n * 3 + 0] - cx;
      const float dy = px[n * 3 + 1] - cy;
      const float dz = px[n * 3 + 2] - cz;
      float dm = fminf(pd[n], dx * dx + dy * dy + dz * dz);
      pd[n] = dm;
      if (dm > bestv) { bestv = dm; besti = n; }  // strictly-greater => lowest n on ties
    }
    s_val[tid] = bestv; s_idx[tid] = besti;
    __syncthreads();
    for (int s = T / 2; s > 0; s >>= 1) {
      if (tid < s) {
        const float ov = s_val[tid + s]; const int oi = s_idx[tid + s];
        if (ov > s_val[tid] || (ov == s_val[tid] && oi < s_idx[tid])) {
          s_val[tid] = ov; s_idx[tid] = oi;
        }
      }
      __syncthreads();
    }
    if (tid == 0) s_far = s_idx[0];
    __syncthreads();
  }
}

__global__ void gather_xyz_kernel(const float* __restrict__ xyz, const int* __restrict__ idx,
                                  float* __restrict__ out, int N, int Q, int total) {
  int t = blockIdx.x * blockDim.x + threadIdx.x;
  if (t >= total) return;                       // total = B*Q*3
  const int c = t % 3; const int q = (t / 3) % Q; const int b = t / (3 * Q);
  out[t] = xyz[((size_t)b * N + idx[b * Q + q]) * 3 + c];
}

// ---------------------------------------------------------------------------
// Ball query: one wave32 per query. Scans points in index order; ballot +
// prefix popcount emits the first `nsample` in-ball indices (== reference's
// sort-then-truncate), padding the tail with the first hit.
// ---------------------------------------------------------------------------
__global__ void ball_query_kernel(const float* __restrict__ xyz, const float* __restrict__ new_xyz,
                                  int* __restrict__ idx, int N, int Q, float r2, int nsample, int BQ) {
  const int lane = threadIdx.x & 31;
  const int wid = blockIdx.x * (blockDim.x >> 5) + (threadIdx.x >> 5);
  if (wid >= BQ) return;
  const int b = wid / Q, q = wid % Q;
  const float* px = xyz + (size_t)b * N * 3;
  const float qx = new_xyz[(b * Q + q) * 3 + 0];
  const float qy = new_xyz[(b * Q + q) * 3 + 1];
  const float qz = new_xyz[(b * Q + q) * 3 + 2];
  int* po = idx + (size_t)(b * Q + q) * nsample;
  int cnt = 0;     // wave-uniform
  int first = 0;   // wave-uniform
  const unsigned long long ltmask = (1ull << lane) - 1ull;
  for (int base = 0; base < N && cnt < nsample; base += 32) {
    const int n = base + lane;
    bool ok = false;
    if (n < N) {
      const float dx = px[n * 3 + 0] - qx;
      const float dy = px[n * 3 + 1] - qy;
      const float dz = px[n * 3 + 2] - qz;
      ok = (dx * dx + dy * dy + dz * dz) <= r2;
    }
    const unsigned long long m = __ballot(ok);
    if (cnt == 0 && m) first = base + (__ffsll((long long)m) - 1);
    if (ok) {
      const int pos = cnt + __popcll(m & ltmask);
      if (pos < nsample) po[pos] = n;
    }
    cnt += __popcll(m);
  }
  if (cnt > nsample) cnt = nsample;
  for (int p = cnt + lane; p < nsample; p += 32) po[p] = first;
}

// ---------------------------------------------------------------------------
// Build grouped features (xyz diff [+ gathered feats]) into f16, K padded.
// feats == nullptr / C == 0 for level 1.
// ---------------------------------------------------------------------------
__global__ void group_kernel(const float* __restrict__ xyz, const float* __restrict__ new_xyz,
                             const int* __restrict__ idx, const half_t* __restrict__ feats,
                             half_t* __restrict__ g, int N, int Q, int S, int C, int Kpad, int total) {
  int r = blockIdx.x * blockDim.x + threadIdx.x;
  if (r >= total) return;                       // total = B*Q*S
  const int s = r % S; const int q = (r / S) % Q; const int b = r / (S * Q);
  const int n = idx[(b * Q + q) * S + s];
  const float* p = xyz + ((size_t)b * N + n) * 3;
  const float* c = new_xyz + (size_t)(b * Q + q) * 3;
  half_t* pg = g + (size_t)r * Kpad;
  pg[0] = (half_t)(p[0] - c[0]);
  pg[1] = (half_t)(p[1] - c[1]);
  pg[2] = (half_t)(p[2] - c[2]);
  if (C > 0) {
    const half_t* pf = feats + ((size_t)b * N + n) * C;
    for (int k = 0; k < C; ++k) pg[3 + k] = pf[k];
  }
  for (int k = 3 + C; k < Kpad; ++k) pg[k] = (half_t)0.f;
}

// level-3 input: concat(l2_xyz[r], l2_pts[r]) padded to Kpad
__global__ void concat3_kernel(const float* __restrict__ l2_xyz, const half_t* __restrict__ l2_pts,
                               half_t* __restrict__ g, int C, int Kpad, int total) {
  int r = blockIdx.x * blockDim.x + threadIdx.x;
  if (r >= total) return;
  half_t* pg = g + (size_t)r * Kpad;
  pg[0] = (half_t)l2_xyz[r * 3 + 0];
  pg[1] = (half_t)l2_xyz[r * 3 + 1];
  pg[2] = (half_t)l2_xyz[r * 3 + 2];
  for (int k = 0; k < C; ++k) pg[3 + k] = l2_pts[(size_t)r * C + k];
  for (int k = 3 + C; k < Kpad; ++k) pg[k] = (half_t)0.f;
}

__global__ void wcvt_kernel(const float* __restrict__ w, half_t* __restrict__ wh,
                            int K, int Kpad, int total) {
  int t = blockIdx.x * blockDim.x + threadIdx.x;
  if (t >= total) return;                       // total = O*Kpad
  const int o = t / Kpad, k = t % Kpad;
  wh[t] = (k < K) ? (half_t)w[o * K + k] : (half_t)0.f;
}

// ---------------------------------------------------------------------------
// WMMA GEMM: C[M,O] = act(A[M,K] * W[O,K]^T + bias), f16 in, f32 accum, f16 out.
// One wave per 16x16 tile, 8 waves per block. K multiple of 32.
// A k order chosen here (contiguous, same for B; not ISA table order): lane<16 row=lane holds K 0..15,
// lane>=16 holds K 16..31. B from W rows: lane<16 -> K 0..15 of col,
// lane>=16 -> K 16..31. D: VGPR r = row (r | r+8), col = lane&15.
// ---------------------------------------------------------------------------
__global__ void __launch_bounds__(256) gemm_wmma_kernel(
    const half_t* __restrict__ A, const half_t* __restrict__ W,
    const float* __restrict__ bias, half_t* __restrict__ C,
    int M, int K, int O, int relu) {
  const int lane  = threadIdx.x & 31;
  const int wave  = threadIdx.x >> 5;
  const int mtile = (blockIdx.y * 8 + wave) * 16;
  const int otile = blockIdx.x * 16;
  const int hl = lane & 15;
  const int hi = lane >> 4;                     // 0 for lanes 0-15, 1 for 16-31
  const half_t* ap = A + (size_t)(mtile + hl) * K + hi * 16;
  const half_t* bp = W + (size_t)(otile + hl) * K + hi * 16;
  v8f acc = {};
  for (int k0 = 0; k0 < K; k0 += 32) {
    union { v16h v; v8h h[2]; } a;
    a.h[0] = *(const v8h*)(ap + k0);            // K = k0 + hi*16 + {0..7}
    a.h[1] = *(const v8h*)(ap + k0 + 8);        // K = k0 + hi*16 + {8..15}
    const v16h bm = *(const v16h*)(bp + k0);    // K = k0 + hi*16 + {0..15}
    acc = __builtin_amdgcn_wmma_f32_16x16x32_f16(false, a.v, false, bm,
                                                 (short)0, acc, false, false);
  }
  const float bv = bias ? bias[otile + hl] : 0.f;
  #pragma unroll
  for (int r = 0; r < 8; ++r) {
    float v = acc[r] + bv;
    if (relu) v = fmaxf(v, 0.f);
    C[(size_t)(mtile + hi * 8 + r) * O + otile + hl] = (half_t)v;
  }
}

__global__ void maxpool_kernel(const half_t* __restrict__ in, half_t* __restrict__ out,
                               int G, int C, int total) {
  int t = blockIdx.x * blockDim.x + threadIdx.x;
  if (t >= total) return;                       // total = Mout*C
  const int c = t % C; const int r = t / C;
  const half_t* p = in + (size_t)r * G * C + c;
  float m = (float)p[0];
  for (int s = 1; s < G; ++s) m = fmaxf(m, (float)p[(size_t)s * C]);
  out[t] = (half_t)m;
}

// x = concat(l1_pts[b*256+q, 0:128], l3[b, 0:512])
__global__ void build_fp_kernel(const half_t* __restrict__ l1p, const half_t* __restrict__ l3,
                                half_t* __restrict__ x, int total) {
  int t = blockIdx.x * blockDim.x + threadIdx.x;
  if (t >= total) return;                       // total = 4096*640
  const int k = t % 640; const int r = t / 640; const int b = r >> 8;
  x[t] = (k < 128) ? l1p[(size_t)r * 128 + k] : l3[(size_t)b * 512 + (k - 128)];
}

// final 64->13 linear in f32 (O=13 too ragged for WMMA, negligible work)
__global__ void logits_kernel(const half_t* __restrict__ h, const float* __restrict__ w,
                              const float* __restrict__ bias, float* __restrict__ out, int total) {
  int t = blockIdx.x * blockDim.x + threadIdx.x;
  if (t >= total) return;                       // total = 4096*13
  const int c = t % 13; const int r = t / 13;
  const half_t* p = h + (size_t)r * 64;
  const float* pw = w + c * 64;
  float acc = bias[c];
  for (int k = 0; k < 64; ++k) acc += (float)p[k] * pw[k];
  out[t] = acc;
}

// softmax over the 13*256 = 3328 values of each batch; output transposed (c*256+n)
__global__ void softmax_kernel(const float* __restrict__ logits, float* __restrict__ out) {
  const int b = blockIdx.x;
  const int tid = threadIdx.x;                  // 256 threads, one per point n
  __shared__ float sred[256];
  const float* pl = logits + (size_t)b * 256 * 13;
  float lmax = -3.4e38f;
  for (int c = 0; c < 13; ++c) lmax = fmaxf(lmax, pl[tid * 13 + c]);
  sred[tid] = lmax; __syncthreads();
  for (int s = 128; s > 0; s >>= 1) { if (tid < s) sred[tid] = fmaxf(sred[tid], sred[tid + s]); __syncthreads(); }
  const float gmax = sred[0]; __syncthreads();
  float lsum = 0.f;
  for (int c = 0; c < 13; ++c) lsum += expf(pl[tid * 13 + c] - gmax);
  sred[tid] = lsum; __syncthreads();
  for (int s = 128; s > 0; s >>= 1) { if (tid < s) sred[tid] += sred[tid + s]; __syncthreads(); }
  const float inv = 1.f / sred[0];
  for (int c = 0; c < 13; ++c)
    out[(size_t)b * 3328 + c * 256 + tid] = expf(pl[tid * 13 + c] - gmax) * inv;
}

static inline int cdiv(int a, int b) { return (a + b - 1) / b; }

extern "C" void kernel_launch(void* const* d_in, const int* in_sizes, int n_in,
                              void* d_out, int out_size, void* d_ws, size_t ws_size,
                              hipStream_t stream) {
  (void)in_sizes; (void)n_in; (void)out_size; (void)ws_size;
  const float* xyz = (const float*)d_in[0];
  const float* w10 = (const float*)d_in[1];  const float* b10 = (const float*)d_in[2];
  const float* w11 = (const float*)d_in[3];  const float* b11 = (const float*)d_in[4];
  const float* w12 = (const float*)d_in[5];  const float* b12 = (const float*)d_in[6];
  const float* w20 = (const float*)d_in[7];  const float* b20 = (const float*)d_in[8];
  const float* w21 = (const float*)d_in[9];  const float* b21 = (const float*)d_in[10];
  const float* w22 = (const float*)d_in[11]; const float* b22 = (const float*)d_in[12];
  const float* w30 = (const float*)d_in[13]; const float* b30 = (const float*)d_in[14];
  const float* w31 = (const float*)d_in[15]; const float* b31 = (const float*)d_in[16];
  const float* w32 = (const float*)d_in[17]; const float* b32 = (const float*)d_in[18];
  const float* fw0 = (const float*)d_in[19]; const float* fb0 = (const float*)d_in[20];
  const float* fw1 = (const float*)d_in[21]; const float* fb1 = (const float*)d_in[22];
  const float* c1w = (const float*)d_in[23]; const float* c1b = (const float*)d_in[24];
  const float* c2w = (const float*)d_in[25]; const float* c2b = (const float*)d_in[26];
  float* out = (float*)d_out;

  const int M1 = B_ * Q1_ * S_;   // 65536
  const int M2 = B_ * Q2_ * S_;   // 16384
  const int M3 = B_ * Q2_;        // 1024
  const int MF = B_ * Q1_;        // 4096

  // -------- workspace bump allocator (256B aligned) --------
  char* wsb = (char*)d_ws; size_t off = 0;
  auto alloc = [&](size_t bytes) -> void* {
    void* p = wsb + off; off += (bytes + 255) & ~(size_t)255; return p;
  };
  float*  dist = (float*)alloc((size_t)B_ * N1_ * 4);
  int*    fps1 = (int*)  alloc((size_t)B_ * Q1_ * 4);
  float*  nx1  = (float*)alloc((size_t)B_ * Q1_ * 3 * 4);
  int*    idx1 = (int*)  alloc((size_t)B_ * Q1_ * S_ * 4);
  half_t* g1   = (half_t*)alloc((size_t)M1 * 32 * 2);
  half_t* h1a  = (half_t*)alloc((size_t)M1 * 64 * 2);
  half_t* h1b  = (half_t*)alloc((size_t)M1 * 64 * 2);
  half_t* h1c  = (half_t*)alloc((size_t)M1 * 128 * 2);
  half_t* l1p  = (half_t*)alloc((size_t)B_ * Q1_ * 128 * 2);
  int*    fps2 = (int*)  alloc((size_t)B_ * Q2_ * 4);
  float*  nx2  = (float*)alloc((size_t)B_ * Q2_ * 3 * 4);
  int*    idx2 = (int*)  alloc((size_t)B_ * Q2_ * S_ * 4);
  half_t* g2   = (half_t*)alloc((size_t)M2 * 160 * 2);
  half_t* h2a  = (half_t*)alloc((size_t)M2 * 128 * 2);
  half_t* h2b  = (half_t*)alloc((size_t)M2 * 128 * 2);
  half_t* h2c  = (half_t*)alloc((size_t)M2 * 256 * 2);
  half_t* l2p  = (half_t*)alloc((size_t)B_ * Q2_ * 256 * 2);
  half_t* g3   = (half_t*)alloc((size_t)M3 * 288 * 2);
  half_t* h3a  = (half_t*)alloc((size_t)M3 * 256 * 2);
  half_t* h3b  = (half_t*)alloc((size_t)M3 * 256 * 2);
  half_t* h3c  = (half_t*)alloc((size_t)M3 * 512 * 2);
  half_t* l3p  = (half_t*)alloc((size_t)B_ * 512 * 2);
  half_t* xfp  = (half_t*)alloc((size_t)MF * 640 * 2);
  half_t* hf0  = (half_t*)alloc((size_t)MF * 256 * 2);
  half_t* hf1  = (half_t*)alloc((size_t)MF * 128 * 2);
  half_t* hc1  = (half_t*)alloc((size_t)MF * 64 * 2);
  float*  lg   = (float*)alloc((size_t)MF * 13 * 4);
  half_t* w10h = (half_t*)alloc((size_t)64 * 32 * 2);
  half_t* w11h = (half_t*)alloc((size_t)64 * 64 * 2);
  half_t* w12h = (half_t*)alloc((size_t)128 * 64 * 2);
  half_t* w20h = (half_t*)alloc((size_t)128 * 160 * 2);
  half_t* w21h = (half_t*)alloc((size_t)128 * 128 * 2);
  half_t* w22h = (half_t*)alloc((size_t)256 * 128 * 2);
  half_t* w30h = (half_t*)alloc((size_t)256 * 288 * 2);
  half_t* w31h = (half_t*)alloc((size_t)256 * 256 * 2);
  half_t* w32h = (half_t*)alloc((size_t)512 * 256 * 2);
  half_t* fw0h = (half_t*)alloc((size_t)256 * 640 * 2);
  half_t* fw1h = (half_t*)alloc((size_t)128 * 256 * 2);
  half_t* c1wh = (half_t*)alloc((size_t)64 * 128 * 2);

  auto wcvt = [&](const float* w, half_t* wh, int O, int K, int Kpad) {
    const int tot = O * Kpad;
    wcvt_kernel<<<cdiv(tot, 256), 256, 0, stream>>>(w, wh, K, Kpad, tot);
  };
  auto gemm = [&](const half_t* A, const half_t* W, const float* bias, half_t* C,
                  int M, int K, int O, int relu) {
    dim3 grid(O / 16, M / 128);
    gemm_wmma_kernel<<<grid, 256, 0, stream>>>(A, W, bias, C, M, K, O, relu);
  };

  // -------- weight conversion to padded f16 --------
  wcvt(w10, w10h, 64, 3, 32);    wcvt(w11, w11h, 64, 64, 64);   wcvt(w12, w12h, 128, 64, 64);
  wcvt(w20, w20h, 128, 131, 160);wcvt(w21, w21h, 128, 128, 128);wcvt(w22, w22h, 256, 128, 128);
  wcvt(w30, w30h, 256, 259, 288);wcvt(w31, w31h, 256, 256, 256);wcvt(w32, w32h, 512, 256, 256);
  wcvt(fw0, fw0h, 256, 640, 640);wcvt(fw1, fw1h, 128, 256, 256);wcvt(c1w, c1wh, 64, 128, 128);

  // -------- SA level 1 --------
  fps_kernel<<<B_, 256, 0, stream>>>(xyz, N1_, Q1_, fps1, dist);
  gather_xyz_kernel<<<cdiv(B_ * Q1_ * 3, 256), 256, 0, stream>>>(xyz, fps1, nx1, N1_, Q1_, B_ * Q1_ * 3);
  ball_query_kernel<<<cdiv(B_ * Q1_ * 32, 256), 256, 0, stream>>>(
      xyz, nx1, idx1, N1_, Q1_, 0.025f * 0.025f, S_, B_ * Q1_);
  group_kernel<<<cdiv(M1, 256), 256, 0, stream>>>(xyz, nx1, idx1, nullptr, g1, N1_, Q1_, S_, 0, 32, M1);
  gemm(g1,  w10h, b10, h1a, M1, 32, 64, 1);
  gemm(h1a, w11h, b11, h1b, M1, 64, 64, 1);
  gemm(h1b, w12h, b12, h1c, M1, 64, 128, 1);
  maxpool_kernel<<<cdiv(B_ * Q1_ * 128, 256), 256, 0, stream>>>(h1c, l1p, S_, 128, B_ * Q1_ * 128);

  // -------- SA level 2 --------
  fps_kernel<<<B_, 256, 0, stream>>>(nx1, Q1_, Q2_, fps2, dist);
  gather_xyz_kernel<<<cdiv(B_ * Q2_ * 3, 256), 256, 0, stream>>>(nx1, fps2, nx2, Q1_, Q2_, B_ * Q2_ * 3);
  ball_query_kernel<<<cdiv(B_ * Q2_ * 32, 256), 256, 0, stream>>>(
      nx1, nx2, idx2, Q1_, Q2_, 0.05f * 0.05f, S_, B_ * Q2_);
  group_kernel<<<cdiv(M2, 256), 256, 0, stream>>>(nx1, nx2, idx2, l1p, g2, Q1_, Q2_, S_, 128, 160, M2);
  gemm(g2,  w20h, b20, h2a, M2, 160, 128, 1);
  gemm(h2a, w21h, b21, h2b, M2, 128, 128, 1);
  gemm(h2b, w22h, b22, h2c, M2, 128, 256, 1);
  maxpool_kernel<<<cdiv(B_ * Q2_ * 256, 256), 256, 0, stream>>>(h2c, l2p, S_, 256, B_ * Q2_ * 256);

  // -------- global level 3 --------
  concat3_kernel<<<cdiv(M3, 256), 256, 0, stream>>>(nx2, l2p, g3, 256, 288, M3);
  gemm(g3,  w30h, b30, h3a, M3, 288, 256, 1);
  gemm(h3a, w31h, b31, h3b, M3, 256, 256, 1);
  gemm(h3b, w32h, b32, h3c, M3, 256, 512, 1);
  maxpool_kernel<<<cdiv(B_ * 512, 256), 256, 0, stream>>>(h3c, l3p, Q2_, 512, B_ * 512);

  // -------- feature propagation + head --------
  build_fp_kernel<<<cdiv(MF * 640, 256), 256, 0, stream>>>(l1p, l3p, xfp, MF * 640);
  gemm(xfp, fw0h, fb0, hf0, MF, 640, 256, 1);
  gemm(hf0, fw1h, fb1, hf1, MF, 256, 128, 1);
  gemm(hf1, c1wh, c1b, hc1, MF, 128, 64, 1);
  logits_kernel<<<cdiv(MF * 13, 256), 256, 0, stream>>>(hc1, c2w, c2b, lg, MF * 13);
  softmax_kernel<<<B_, 256, 0, stream>>>(lg, out);
}
